// LinearAttention_13348758356175
// MI455X (gfx1250) — matrix operand order repaired and checked by reading, not compiled, not run on hardware
//
#include <hip/hip_runtime.h>
#include <hip/hip_bf16.h>

typedef __attribute__((ext_vector_type(16))) __bf16 v16bf;
typedef __attribute__((ext_vector_type(8)))  float  v8f;

#define B_  2
#define S_  2048
#define H_  1024
#define NH_ 16
#define HD_ 64
#define CS_ 512
#define NC_ 4

#if __has_builtin(__builtin_amdgcn_global_load_async_to_lds_b128)
#define HAS_ASYNC 1
#else
#define HAS_ASYNC 0
#endif

// vector-of-4-int pointee types matching the async-copy builtin's signature
typedef int v4i_ __attribute__((vector_size(16)));
typedef __attribute__((address_space(1))) v4i_ gv4i;   // global
typedef __attribute__((address_space(3))) v4i_ lv4i;   // LDS

static __device__ __forceinline__ unsigned short f2bf(float f) {
  union { float f; unsigned int u; } x; x.f = f;
  unsigned int u = x.u + 0x7FFFu + ((x.u >> 16) & 1u);   // round-to-nearest-even
  return (unsigned short)(u >> 16);
}

// Load 16 bf16 elements (two 16B chunks) into a WMMA A/B operand register set.
static __device__ __forceinline__ v16bf ldab(const unsigned short* p0, const unsigned short* p1) {
  union { v16bf v; uint4 q[2]; } r;
  r.q[0] = *(const uint4*)p0;
  r.q[1] = *(const uint4*)p1;
  return r.v;
}

static __device__ __forceinline__ v8f wmma_bf16(v16bf a, v16bf b, v8f c) {
  return __builtin_amdgcn_wmma_f32_16x16x32_bf16(false, a, false, b, (short)0, c, false, false);
}

// 16-byte global -> LDS copy.  Async (no VGPR staging, ASYNCcnt-tracked) when
// the toolchain exposes the gfx1250 builtin; plain copy otherwise.
static __device__ __forceinline__ void async_cp16(unsigned short* l, const unsigned short* g) {
#if HAS_ASYNC
  __builtin_amdgcn_global_load_async_to_lds_b128((gv4i*)g, (lv4i*)l, 0, 0);
#else
  *(uint4*)l = *(const uint4*)g;
#endif
}

static __device__ __forceinline__ void wait_async6() {
#if HAS_ASYNC
#if __has_builtin(__builtin_amdgcn_s_wait_asynccnt)
  __builtin_amdgcn_s_wait_asynccnt(6);
#else
  asm volatile("s_wait_asynccnt 0x6" ::: "memory");
#endif
#endif
}
static __device__ __forceinline__ void wait_async0() {
#if HAS_ASYNC
#if __has_builtin(__builtin_amdgcn_s_wait_asynccnt)
  __builtin_amdgcn_s_wait_asynccnt(0);
#else
  asm volatile("s_wait_asynccnt 0x0" ::: "memory");
#endif
#endif
}

// ---------------- precision conversion / weight transpose ----------------
__global__ void k_cvt(const float* __restrict__ src, unsigned short* __restrict__ dst, int n) {
  int i = blockIdx.x * blockDim.x + threadIdx.x;
  if (i < n) dst[i] = f2bf(src[i]);
}

// dst[n][k] = bf16(src[k][n]);  src is K x N row-major, dst is N x K row-major
__global__ void k_transpose_cvt(const float* __restrict__ src, unsigned short* __restrict__ dst,
                                int K, int N) {
  int i = blockIdx.x * blockDim.x + threadIdx.x;
  if (i >= K * N) return;
  int n = i / K, k = i % K;
  dst[i] = f2bf(src[(size_t)k * N + n]);
}

// ======================================================================
// Async-copy double-buffered GEMM body: block tile 128(M) x 64(N), K in
// 16 stages of 64.  8 waves split M; B is fetched from global once per
// block.  Stage st+2's global->LDS async copies are issued right after
// the compute barrier of stage st; the loop head waits ASYNCcnt<=6 (one
// stage outstanding) then barriers.  No VGPR staging -> no spills.
// ======================================================================
#define GEMM_ISSUE(stg, buf) do {                                               \
    _Pragma("unroll")                                                           \
    for (int t = 0; t < 4; ++t)                                                 \
      async_cp16(&As[buf][adst[t]], asrc[t] + (size_t)(stg) * 64);              \
    _Pragma("unroll")                                                           \
    for (int t = 0; t < 2; ++t)                                                 \
      async_cp16(&Bs[buf][bdst[t]], bsrc[t] + (size_t)(stg) * 64);              \
  } while (0)

#define GEMM_BODY(Aptr, Bptr, KSTEPS)                                           \
  __shared__ unsigned short As[2][128 * 64];                                    \
  __shared__ unsigned short Bs[2][64 * 64];                                     \
  const int tid = threadIdx.x;                                                  \
  const int wave = tid >> 5, lane = tid & 31;                                   \
  const int l16 = lane & 15, lh = lane >> 4;                                    \
  const unsigned short* asrc[4]; int adst[4];                                   \
  const unsigned short* bsrc[2]; int bdst[2];                                   \
  _Pragma("unroll")                                                             \
  for (int t = 0; t < 4; ++t) {                                                 \
    int idx = tid + t * 256, row = idx >> 3, c = idx & 7;                       \
    asrc[t] = Aptr + (size_t)row * 1024 + c * 8;                                \
    adst[t] = row * 64 + c * 8;                                                 \
  }                                                                             \
  _Pragma("unroll")                                                             \
  for (int t = 0; t < 2; ++t) {                                                 \
    int idx = tid + t * 256, row = idx >> 3, c = idx & 7;                       \
    bsrc[t] = Bptr + (size_t)row * 1024 + c * 8;                                \
    bdst[t] = row * 64 + c * 8;                                                 \
  }                                                                             \
  GEMM_ISSUE(0, 0);                                                             \
  GEMM_ISSUE(1, 1);                                                             \
  v8f acc[4] = {};                                                              \
  const int arow = (wave * 16 + l16) * 64;                                      \
  for (int st = 0; st < (KSTEPS); ++st) {                                       \
    if (st + 1 < (KSTEPS)) wait_async6(); else wait_async0();                   \
    __syncthreads();                                                            \
    const unsigned short* Al = As[st & 1];                                      \
    const unsigned short* Bl = Bs[st & 1];                                      \
    v16bf a0 = ldab(Al + arow + lh * 16,     Al + arow + lh * 16 + 8);          \
    v16bf a1 = ldab(Al + arow + 32 + lh * 16, Al + arow + 40 + lh * 16);        \
    v16bf bt[4];                                                                \
    _Pragma("unroll")                                                           \
    for (int nt = 0; nt < 4; ++nt) {                                            \
      const unsigned short* br = Bl + (nt * 16 + l16) * 64;                     \
      bt[nt] = ldab(br + lh * 16, br + lh * 16 + 8);                            \
    }                                                                           \
    _Pragma("unroll")                                                           \
    for (int nt = 0; nt < 4; ++nt) acc[nt] = wmma_bf16(a0, bt[nt], acc[nt]);    \
    _Pragma("unroll")                                                           \
    for (int nt = 0; nt < 4; ++nt) {                                            \
      const unsigned short* br = Bl + (nt * 16 + l16) * 64;                     \
      bt[nt] = ldab(br + 32 + lh * 16, br + 40 + lh * 16);                      \
    }                                                                           \
    _Pragma("unroll")                                                           \
    for (int nt = 0; nt < 4; ++nt) acc[nt] = wmma_bf16(a1, bt[nt], acc[nt]);    \
    __syncthreads();                                                            \
    if (st + 2 < (KSTEPS)) { GEMM_ISSUE(st + 2, st & 1); }                      \
  }

// ---------------- QKV GEMM: [4096,1024] x [1024,3072], scatter to q/k/vT ----------------
__global__ __launch_bounds__(256, 1) void k_qkv(const unsigned short* __restrict__ xb,
                                                const unsigned short* __restrict__ wqT,
                                                const float* __restrict__ bq,
                                                unsigned short* __restrict__ qb,
                                                unsigned short* __restrict__ kb,
                                                unsigned short* __restrict__ vtb) {
  const int m0 = (blockIdx.x / 48) * 128;          // 32 M-blocks
  const int n0 = (blockIdx.x % 48) * 64;           // 48 N-blocks
  const unsigned short* Ag = xb + (size_t)m0 * 1024;
  const unsigned short* Bg = wqT + (size_t)n0 * 1024;
  GEMM_BODY(Ag, Bg, 16)
#pragma unroll
  for (int nt = 0; nt < 4; ++nt) {
    int n = n0 + nt * 16 + l16;
    float bias = bq[n];
    int t = n / H_;
    int rem = n % H_;
    int hh = rem / HD_, d = rem % HD_;
#pragma unroll
    for (int r = 0; r < 8; ++r) {
      int m = m0 + wave * 16 + r + lh * 8;
      int bb = m / S_, srow = m % S_;
      float val = acc[nt][r] + bias;
      if (t == 0)      qb[((size_t)(bb * NH_ + hh) * S_ + srow) * HD_ + d] = f2bf(val * 0.125f);
      else if (t == 1) kb[((size_t)(bb * NH_ + hh) * S_ + srow) * HD_ + d] = f2bf(val);
      else             vtb[((size_t)(bb * NH_ + hh) * HD_ + d) * S_ + srow] = f2bf(val);
    }
  }
}

// ---------------- output projection: [4096,1024] x [1024,1024] + bias -> f32 ----------------
__global__ __launch_bounds__(256, 1) void k_proj(const unsigned short* __restrict__ abf,
                                                 const unsigned short* __restrict__ wpT,
                                                 const float* __restrict__ bp,
                                                 float* __restrict__ out) {
  const int m0 = (blockIdx.x / 16) * 128;          // 32 M-blocks
  const int n0 = (blockIdx.x % 16) * 64;           // 16 N-blocks
  const unsigned short* Ag = abf + (size_t)m0 * 1024;
  const unsigned short* Bg = wpT + (size_t)n0 * 1024;
  GEMM_BODY(Ag, Bg, 16)
#pragma unroll
  for (int nt = 0; nt < 4; ++nt) {
    int n = n0 + nt * 16 + l16;
    float bias = bp[n];
#pragma unroll
    for (int r = 0; r < 8; ++r) {
      int m = m0 + wave * 16 + r + lh * 8;
      out[(size_t)m * H_ + n] = acc[nt][r] + bias;
    }
  }
}

// ---------------- chunk-local attention ----------------
__global__ __launch_bounds__(256, 1) void k_attn(const unsigned short* __restrict__ qb,
                                                 const unsigned short* __restrict__ kb,
                                                 const unsigned short* __restrict__ vtb,
                                                 unsigned short* __restrict__ ab) {
  __shared__ unsigned short kch[128 * 64];    // 16 KB: 128 k-rows x 64 hd
  __shared__ unsigned short vtch[64 * 128];   // 16 KB: 64 hd x 128 k-rows (transposed v)
  __shared__ unsigned short pbuf[8 * 512];    //  8 KB: per-wave 16x32 probability staging
  const int wave = threadIdx.x >> 5, lane = threadIdx.x & 31;
  const int l16 = lane & 15, lh = lane >> 4;
  const int bid = blockIdx.x;
  const int p  = bid & 3;          // 128-row part of q chunk
  const int ic = (bid >> 2) & 3;   // q chunk index
  const int h  = (bid >> 4) & 15;
  const int b  = bid >> 8;
  const size_t head = (size_t)(b * NH_ + h);
  const int qrow0 = ic * CS_ + p * 128 + wave * 16;

  const unsigned short* qrow = qb + (head * S_ + qrow0 + l16) * HD_;
  v16bf aq0 = ldab(qrow + lh * 16,     qrow + lh * 16 + 8);   // K = 0..31 of hd
  v16bf aq1 = ldab(qrow + 32 + lh * 16, qrow + 40 + lh * 16); // K = 32..63 of hd

  const unsigned short* kbase = kb + head * S_ * HD_;
  const unsigned short* vbase = vtb + head * HD_ * S_;

  v8f tot[4] = {};
  for (int j = 0; j < NC_; ++j) {
    v8f U[4] = {};
    float rs[8] = {};
    for (int sub = 0; sub < 4; ++sub) {                       // 128 k-rows per sub
      __syncthreads();
      const unsigned short* ksrc = kbase + (size_t)(j * CS_ + sub * 128) * HD_;
#pragma unroll
      for (int t = 0; t < 4; ++t) {
        int idx = threadIdx.x + t * 256;                      // 1024 uint4 chunks
        ((uint4*)kch)[idx] = ((const uint4*)ksrc)[idx];
      }
      const unsigned short* vsrc = vbase + j * CS_ + sub * 128;
#pragma unroll
      for (int t = 0; t < 4; ++t) {
        int idx = threadIdx.x + t * 256;
        int row = idx >> 4, c8 = idx & 15;
        ((uint4*)vtch)[(row << 4) + c8] = ((const uint4*)(vsrc + (size_t)row * S_))[c8];
      }
      // prefetch the next sub-tile (or next chunk's first sub-tile) into cache
      // while this one is being consumed -> global_prefetch_b8
      {
        int nj = j, ns = sub + 1;
        if (ns == 4) { ns = 0; nj = j + 1; }
        if (nj < NC_) {
          const unsigned short* kpre = kbase + (size_t)(nj * CS_ + ns * 128) * HD_;
          __builtin_prefetch(kpre + (size_t)threadIdx.x * 32, 0, 0);   // 16 KB, 64B/thread
          const unsigned short* vpre = vbase + nj * CS_ + ns * 128;
          __builtin_prefetch(vpre + (size_t)(threadIdx.x >> 2) * S_ + (threadIdx.x & 3) * 32, 0, 0);
        }
      }
      __syncthreads();
#pragma unroll
      for (int pair = 0; pair < 4; ++pair) {                  // 32 k-cols per pair
#pragma unroll
        for (int half = 0; half < 2; ++half) {
          const int ct = pair * 2 + half;
          const unsigned short* krow = kch + (ct * 16 + l16) * 64;
          v16bf bk0 = ldab(krow + lh * 16,      krow + lh * 16 + 8);
          v16bf bk1 = ldab(krow + 32 + lh * 16, krow + 40 + lh * 16);
          v8f s = {};
          s = wmma_bf16(aq0, bk0, s);
          s = wmma_bf16(aq1, bk1, s);
#pragma unroll
          for (int r = 0; r < 8; ++r) {                       // exp + row-sum + stage P
            float e = __expf(s[r]);
            rs[r] += e;
            pbuf[wave * 512 + (r + lh * 8) * 32 + half * 16 + l16] = f2bf(e);
          }
        }
        asm volatile("s_wait_dscnt 0x0" ::: "memory");        // P stores visible to loads
        const unsigned short* prow = pbuf + wave * 512 + l16 * 32;
        v16bf ap = ldab(prow + lh * 16, prow + lh * 16 + 8);   // P in A-matrix layout
#pragma unroll
        for (int nt = 0; nt < 4; ++nt) {
          const unsigned short* vrow = vtch + (nt * 16 + l16) * 128 + pair * 32;
          v16bf bv = ldab(vrow + lh * 16, vrow + lh * 16 + 8);
          U[nt] = wmma_bf16(ap, bv, U[nt]);
        }
      }
    }
    // finish chunk-local softmax: reduce row sums across the 16-lane half, normalize,
    // and accumulate the 1/nc-averaged contribution of this k-chunk.
#pragma unroll
    for (int r = 0; r < 8; ++r) {
      float v = rs[r];
      v += __shfl_xor(v, 1, 32);
      v += __shfl_xor(v, 2, 32);
      v += __shfl_xor(v, 4, 32);
      v += __shfl_xor(v, 8, 32);
      float inv = 0.25f / v;                                  // 1 / (nc * rowsum)
#pragma unroll
      for (int nt = 0; nt < 4; ++nt) tot[nt][r] += U[nt][r] * inv;
    }
  }
#pragma unroll
  for (int nt = 0; nt < 4; ++nt) {
#pragma unroll
    for (int r = 0; r < 8; ++r) {
      int srow = qrow0 + r + lh * 8;
      ab[((size_t)(b * S_ + srow)) * H_ + h * HD_ + nt * 16 + l16] = f2bf(tot[nt][r]);
    }
  }
}

extern "C" void kernel_launch(void* const* d_in, const int* in_sizes, int n_in,
                              void* d_out, int out_size, void* d_ws, size_t ws_size,
                              hipStream_t stream) {
  const float* x  = (const float*)d_in[0];
  const float* wq = (const float*)d_in[1];
  const float* bq = (const float*)d_in[2];
  const float* wp = (const float*)d_in[3];
  const float* bp = (const float*)d_in[4];
  float* out = (float*)d_out;

  char* ws = (char*)d_ws;
  unsigned short* xb  = (unsigned short*)(ws);                       // 8 MiB  x bf16
  unsigned short* wqT = (unsigned short*)(ws + (size_t)( 8u << 20)); // 6 MiB  w_qkv^T bf16
  unsigned short* wpT = (unsigned short*)(ws + (size_t)(14u << 20)); // 2 MiB  w_proj^T bf16
  unsigned short* qb  = (unsigned short*)(ws + (size_t)(16u << 20)); // 8 MiB  q [B,nh,S,hd]
  unsigned short* kb  = (unsigned short*)(ws + (size_t)(24u << 20)); // 8 MiB  k [B,nh,S,hd]
  unsigned short* vtb = (unsigned short*)(ws + (size_t)(32u << 20)); // 8 MiB  v^T [B,nh,hd,S]
  unsigned short* ab  = (unsigned short*)(ws + (size_t)(40u << 20)); // 8 MiB  attn out bf16

  const int nx = B_ * S_ * H_;  // 4194304
  k_cvt<<<(nx + 255) / 256, 256, 0, stream>>>(x, xb, nx);
  k_transpose_cvt<<<(H_ * 3 * H_ + 255) / 256, 256, 0, stream>>>(wq, wqT, H_, 3 * H_);
  k_transpose_cvt<<<(H_ * H_ + 255) / 256, 256, 0, stream>>>(wp, wpT, H_, H_);
  k_qkv<<<1536, 256, 0, stream>>>(xb, wqT, bq, qb, kb, vtb);
  k_attn<<<512, 256, 0, stream>>>(qb, kb, vtb, ab);
  k_proj<<<512, 256, 0, stream>>>(ab, wpT, bp, out);
}
